// SubLSTMCudaCell_64072322122354
// MI455X (gfx1250) — matrix operand order repaired and checked by reading, not compiled, not run on hardware
//
#include <hip/hip_runtime.h>

typedef __bf16 bf16_t;
typedef bf16_t v16bf __attribute__((ext_vector_type(16)));
typedef bf16_t v4bf  __attribute__((ext_vector_type(4)));
typedef float  v8f   __attribute__((ext_vector_type(8)));

#define B_ROWS 8192
#define K_DIM  2048
#define H_DIM  1024
#define FOURH  4096

#define BM 128
#define BN 16
#define BK 32
#define NKT (K_DIM / BK)
#define LDA 40   // padded k-stride (elements) for A tile
#define LDB 40   // padded k-stride (elements) for B tile

__device__ __forceinline__ float sigmoidf_fast(float x) {
  return 1.0f / (1.0f + __expf(-x));
}

// Gather one K-tile (A: 128x32 from old_h|input, B: 4 gates x 32x16 from W) into registers.
__device__ __forceinline__ void stage_load(const float* __restrict__ input,
                                           const float* __restrict__ old_h,
                                           const float* __restrict__ W,
                                           int blockRow, int blockCol, int tid, int kt,
                                           float4 aV[4], float4 bV[2]) {
  const int k0 = kt * BK;
  // K tiles of 32 never straddle the [old_h | input] boundary at 1024
  const float* src;
  int scol;
  if (k0 < H_DIM) { src = old_h; scol = k0; }
  else            { src = input; scol = k0 - H_DIM; }

  const int kk = (tid & 7) * 4;   // 0,4,...,28
  const int r0 = tid >> 3;        // 0..31
  #pragma unroll
  for (int p = 0; p < 4; ++p) {
    aV[p] = *reinterpret_cast<const float4*>(
        src + (size_t)(blockRow + r0 + p * 32) * H_DIM + scol + kk);
  }

  const int g  = tid >> 6;        // gate 0..3
  const int t2 = tid & 63;
  const int n0 = (t2 & 3) * 4;    // 0,4,8,12
  const int kb = t2 >> 2;         // 0..15
  #pragma unroll
  for (int p = 0; p < 2; ++p) {
    bV[p] = *reinterpret_cast<const float4*>(
        W + (size_t)(k0 + kb + p * 16) * FOURH + g * H_DIM + blockCol + n0);
  }
}

// Convert to bf16 (hardware cvt) and store into the LDS tile buffers.
__device__ __forceinline__ void stage_store(bf16_t (*Als)[LDA],           // [BM][LDA]
                                            bf16_t (*Bls)[BN][LDB],      // [4][BN][LDB]
                                            int tid, const float4 aV[4], const float4 bV[2]) {
  const int kk = (tid & 7) * 4;
  const int r0 = tid >> 3;
  #pragma unroll
  for (int p = 0; p < 4; ++p) {
    v4bf pk = {(bf16_t)aV[p].x, (bf16_t)aV[p].y, (bf16_t)aV[p].z, (bf16_t)aV[p].w};
    *reinterpret_cast<v4bf*>(&Als[r0 + p * 32][kk]) = pk;
  }

  const int g  = tid >> 6;
  const int t2 = tid & 63;
  const int n0 = (t2 & 3) * 4;
  const int kb = t2 >> 2;
  #pragma unroll
  for (int p = 0; p < 2; ++p) {
    const int k = kb + p * 16;
    Bls[g][n0 + 0][k] = (bf16_t)bV[p].x;   // transposed store: [n][k]
    Bls[g][n0 + 1][k] = (bf16_t)bV[p].y;
    Bls[g][n0 + 2][k] = (bf16_t)bV[p].z;
    Bls[g][n0 + 3][k] = (bf16_t)bV[p].w;
  }
}

__launch_bounds__(256)
__global__ void sublstm_wmma_kernel(const float* __restrict__ input,
                                    const float* __restrict__ old_h,
                                    const float* __restrict__ old_cell,
                                    const float* __restrict__ W,     // [K_DIM, FOURH]
                                    const float* __restrict__ bias,  // [FOURH]
                                    float* __restrict__ out) {       // [2 * B * H]
  __shared__ bf16_t Als[2][BM][LDA];        // double-buffered A tile [row][k]
  __shared__ bf16_t Bls[2][4][BN][LDB];     // double-buffered B tiles, transposed [gate][n][k]

  const int tid  = threadIdx.x;
  const int lane = tid & 31;
  const int wave = tid >> 5;
  const int half = lane >> 4;
  const int l16  = lane & 15;

  const int blockRow = blockIdx.y * BM;
  const int blockCol = blockIdx.x * BN;

  const v8f vzero = {0.f, 0.f, 0.f, 0.f, 0.f, 0.f, 0.f, 0.f};
  v8f acc[4];
  #pragma unroll
  for (int g = 0; g < 4; ++g) acc[g] = vzero;

  // ---- Prologue: stage tile 0 ----
  {
    float4 aV[4], bV[2];
    stage_load(input, old_h, W, blockRow, blockCol, tid, 0, aV, bV);
    stage_store(Als[0], Bls[0], tid, aV, bV);
  }
  __syncthreads();

  for (int kt = 0; kt < NKT; ++kt) {
    const int cur = kt & 1;
    const int nxt = cur ^ 1;
    const bool have_next = (kt + 1 < NKT);

    // ---- Issue next tile's global loads first (overlap with WMMA) ----
    float4 aV[4], bV[2];
    if (have_next) {
      stage_load(input, old_h, W, blockRow, blockCol, tid, kt + 1, aV, bV);
    }
    if (kt + 2 < NKT) {
      // prefetch tile kt+2 into L2 (global_prefetch_b8)
      const int k0p = (kt + 2) * BK;
      const float* srcp;
      int scolp;
      if (k0p < H_DIM) { srcp = old_h; scolp = k0p; }
      else             { srcp = input; scolp = k0p - H_DIM; }
      __builtin_prefetch(srcp + (size_t)(blockRow + (tid >> 3)) * H_DIM + scolp + (tid & 7) * 4, 0, 1);
      __builtin_prefetch(W + (size_t)(k0p + ((tid & 63) >> 2)) * FOURH + (tid >> 6) * H_DIM + blockCol, 0, 1);
    }

    // ---- A fragment: 16x32 bf16 per wave (ISA 7.12.2 16-bit A layout) ----
    v16bf afrag;
    {
      const int arow = wave * 16 + l16;
      const int kb = half * 8;   // lanes 0-15: K 0-7/16-23 ; lanes 16-31: K 8-15/24-31
      uint4* ap = reinterpret_cast<uint4*>(&afrag);
      ap[0] = *reinterpret_cast<const uint4*>(&Als[cur][arow][kb]);
      ap[1] = *reinterpret_cast<const uint4*>(&Als[cur][arow][kb + 16]);
    }

    // ---- 4 gate WMMAs sharing one A fragment ----
    #pragma unroll
    for (int g = 0; g < 4; ++g) {
      v16bf bfrag;
      const int kb = half * 8;   // lanes 0-15: K 0-7/16-23 ; lanes 16-31: K 8-15/24-31
      uint4* bp = reinterpret_cast<uint4*>(&bfrag);
      bp[0] = *reinterpret_cast<const uint4*>(&Bls[cur][g][l16][kb]);
      bp[1] = *reinterpret_cast<const uint4*>(&Bls[cur][g][l16][kb + 16]);
      acc[g] = __builtin_amdgcn_wmma_f32_16x16x32_bf16(
          false, afrag, false, bfrag, (short)0, acc[g], false, false);
    }

    // ---- Convert + store next tile into the other buffer ----
    if (have_next) {
      stage_store(Als[nxt], Bls[nxt], tid, aV, bV);
    }
    __syncthreads();
  }

  // ---- Fused subLSTM epilogue (no gate matrix ever written) ----
  const int gcol = blockCol + l16;
  float bg[4];
  #pragma unroll
  for (int g = 0; g < 4; ++g) bg[g] = bias[g * H_DIM + gcol];

  const int rowBase = blockRow + wave * 16 + half * 8;  // C/D layout: VGPR r -> M = r (+8 hi half)
  #pragma unroll
  for (int r = 0; r < 8; ++r) {
    const size_t idx = (size_t)(rowBase + r) * H_DIM + gcol;
    const float si = sigmoidf_fast(acc[0][r] + bg[0]);
    const float sf = sigmoidf_fast(acc[1][r] + bg[1]);
    const float sz = sigmoidf_fast(acc[2][r] + bg[2]);
    const float so = sigmoidf_fast(acc[3][r] + bg[3]);
    const float oc = old_cell[idx];
    const float nc = sf * oc + sz - si;            // subtractive gating
    const float nh = sigmoidf_fast(nc) - so;
    out[idx] = nh;                                 // new_h
    out[(size_t)B_ROWS * H_DIM + idx] = nc;        // new_cell
  }
}

extern "C" void kernel_launch(void* const* d_in, const int* in_sizes, int n_in,
                              void* d_out, int out_size, void* d_ws, size_t ws_size,
                              hipStream_t stream) {
  const float* input    = (const float*)d_in[0];
  const float* old_h    = (const float*)d_in[1];
  const float* old_cell = (const float*)d_in[2];
  const float* W        = (const float*)d_in[3];
  const float* bias     = (const float*)d_in[4];
  float* out = (float*)d_out;

  dim3 grid(H_DIM / BN, B_ROWS / BM);   // (64, 64)
  sublstm_wmma_kernel<<<grid, 256, 0, stream>>>(input, old_h, old_cell, W, bias, out);
}
